// Qwen35HybridModel_50208167690540
// MI455X (gfx1250) — matrix operand order repaired and checked by reading, not compiled, not run on hardware
//
#include <hip/hip_runtime.h>
#include <math.h>

// ---------------------------------------------------------------------------
// Types for CDNA5 WMMA (wave32)
// ---------------------------------------------------------------------------
typedef __attribute__((ext_vector_type(16))) __bf16 v16bf;
typedef __attribute__((ext_vector_type(8)))  __bf16 v8bf;
typedef __attribute__((ext_vector_type(4)))  __bf16 v4bf;
typedef __attribute__((ext_vector_type(8)))  float  v8f;
typedef __attribute__((ext_vector_type(4)))  float  v4f;

static constexpr int BM = 128, BN = 128, BK = 32, LDSP = BK + 8; // +8 bf16 pad

// Native conversion: lowers to v_cvt_pk_bf16_f32 when vectorized.
__device__ __forceinline__ __bf16 f2bf(float f) { return (__bf16)f; }

__device__ __forceinline__ v16bf cat16(v8bf lo, v8bf hi) {
  return __builtin_shufflevector(lo, hi, 0,1,2,3,4,5,6,7,8,9,10,11,12,13,14,15);
}

__device__ __forceinline__ float siluf(float x) { return x / (1.f + expf(-x)); }

// ---------------------------------------------------------------------------
// Generic batched WMMA GEMM:  C = alpha * A @ B(^T)
//   A: [M,K] f32, row-major, lda, batch offset z*strideA
//   B: non-trans [K,N] ldb / trans [N,K] ldb, batch offset (z/divB)*strideB
//   C: [M,N] f32 ldc, batch offset z*strideC
// 8 waves; each wave computes a 32x64 slab = 2x4 WMMA 16x16 tiles.
// ---------------------------------------------------------------------------
template<bool TRANSB>
__global__ __launch_bounds__(256) void gemm_wmma_kernel(
    const float* __restrict__ A, const float* __restrict__ B, float* __restrict__ C,
    int M, int N, int K, int lda, int ldb, int ldc,
    long long strideA, long long strideB, long long strideC,
    int divB, float alpha)
{
  __shared__ __bf16 sA[BM][LDSP];
  __shared__ __bf16 sB[BN][LDSP];   // stored K-contiguous per n-row

  const int z = blockIdx.z;
  A += (long long)z * strideA;
  B += (long long)(z / divB) * strideB;
  C += (long long)z * strideC;

  const int bm = blockIdx.y * BM;
  const int bn = blockIdx.x * BN;

  const int tid  = threadIdx.x;
  const int lane = tid & 31;
  const int wid  = tid >> 5;
  const int wm   = wid & 3;      // 4 row-slabs of 32
  const int wn   = wid >> 2;     // 2 col-slabs of 64
  const int l16  = lane & 15;
  const int half = lane >> 4;

  const v8f vzero = {0.f,0.f,0.f,0.f,0.f,0.f,0.f,0.f};
  v8f acc[2][4];
  #pragma unroll
  for (int i = 0; i < 2; ++i)
    #pragma unroll
    for (int j = 0; j < 4; ++j) acc[i][j] = vzero;

  for (int k0 = 0; k0 < K; k0 += BK) {
    // --- prefetch next K tile (global_prefetch_b8) ---
    if (k0 + BK < K) {
      int pr = bm + (tid & 127);
      if (pr < M) __builtin_prefetch(A + (size_t)pr * lda + k0 + BK, 0, 1);
      if (TRANSB) {
        int pn = bn + (tid & 127);
        if (pn < N) __builtin_prefetch(B + (size_t)pn * ldb + k0 + BK, 0, 1);
      } else if (tid < 32 && (k0 + BK + lane) < K) {
        __builtin_prefetch(B + (size_t)(k0 + BK + lane) * ldb + bn, 0, 1);
      }
    }

    // --- A tile: 128x32 f32 -> bf16 (packed cvt + 64-bit LDS store) ---
    #pragma unroll
    for (int r = 0; r < 4; ++r) {
      int f   = tid + r * 256;        // 0..1023
      int row = f >> 3;               // 8 float4 per row
      int c4  = (f & 7) << 2;
      v4f v = {0.f, 0.f, 0.f, 0.f};
      int gm = bm + row;
      if (gm < M) v = *(const v4f*)(A + (size_t)gm * lda + k0 + c4);
      *(v4bf*)&sA[row][c4] = __builtin_convertvector(v, v4bf);
    }
    // --- B tile -> sB[n][k] (K-contiguous) ---
    if (TRANSB) {
      #pragma unroll
      for (int r = 0; r < 4; ++r) {
        int f   = tid + r * 256;
        int row = f >> 3;             // n
        int c4  = (f & 7) << 2;       // k
        v4f v = {0.f, 0.f, 0.f, 0.f};
        int gn = bn + row;
        if (gn < N) v = *(const v4f*)(B + (size_t)gn * ldb + k0 + c4);
        *(v4bf*)&sB[row][c4] = __builtin_convertvector(v, v4bf);
      }
    } else {
      #pragma unroll
      for (int r = 0; r < 4; ++r) {
        int f    = tid + r * 256;
        int krow = f >> 5;            // 32 float4 per k-row
        int c4   = (f & 31) << 2;     // n offset
        int gn0  = bn + c4;
        const float* src = B + (size_t)(k0 + krow) * ldb + gn0;
        v4f v = {0.f, 0.f, 0.f, 0.f};
        if (gn0 + 3 < N) {
          v = *(const v4f*)src;
        } else {
          if (gn0 + 0 < N) v.x = src[0];
          if (gn0 + 1 < N) v.y = src[1];
          if (gn0 + 2 < N) v.z = src[2];
          if (gn0 + 3 < N) v.w = src[3];
        }
        sB[c4 + 0][krow] = f2bf(v.x);   // transpose scatter: scalar stores
        sB[c4 + 1][krow] = f2bf(v.y);
        sB[c4 + 2][krow] = f2bf(v.z);
        sB[c4 + 3][krow] = f2bf(v.w);
      }
    }
    __syncthreads();

    // --- fragments per the CDNA5 VGPR layouts ---
    v16bf af[2], bfv[4];
    #pragma unroll
    for (int i = 0; i < 2; ++i) {
      // A: lanes0-15 K0..7 | K16..23 ; lanes16-31 K8..15 | K24..31
      const __bf16* p = &sA[wm * 32 + i * 16 + l16][half * 8];
      af[i] = cat16(*(const v8bf*)p, *(const v8bf*)(p + 16));
    }
    #pragma unroll
    for (int j = 0; j < 4; ++j) {
      // B: lanes0-15 K0..7 | K16..23 ; lanes16-31 K8..15 | K24..31
      const __bf16* p = &sB[wn * 64 + j * 16 + l16][half * 8];
      bfv[j] = cat16(*(const v8bf*)p, *(const v8bf*)(p + 16));
    }
    #pragma unroll
    for (int i = 0; i < 2; ++i)
      #pragma unroll
      for (int j = 0; j < 4; ++j)
        acc[i][j] = __builtin_amdgcn_wmma_f32_16x16x32_bf16(
            false, af[i], false, bfv[j], (short)0, acc[i][j], false, false);
    __syncthreads();
  }

  // --- store C (VGPR e: M = e + 8*half, N = l16) ---
  #pragma unroll
  for (int i = 0; i < 2; ++i) {
    int mb = bm + wm * 32 + i * 16 + half * 8;
    #pragma unroll
    for (int j = 0; j < 4; ++j) {
      int n = bn + wn * 64 + j * 16 + l16;
      if (n < N) {
        #pragma unroll
        for (int e = 0; e < 8; ++e) {
          int m = mb + e;
          if (m < M) C[(size_t)m * ldc + n] = alpha * acc[i][j][e];
        }
      }
    }
  }
}

// ---------------------------------------------------------------------------
// RMSNorm: rows x C, input offset = (r/nh)*ld + (r%nh)*C, compact output.
// expw!=0 applies w_eff = exp(w) (head norm).
// ---------------------------------------------------------------------------
__global__ __launch_bounds__(256) void rmsnorm_kernel(
    const float* __restrict__ x, const float* __restrict__ w, float* __restrict__ y,
    int C, int ld, int nh, int expw, float eps)
{
  int r = blockIdx.x;
  const float* xr = x + (size_t)(r / nh) * ld + (size_t)(r % nh) * C;
  float* yr = y + (size_t)r * C;
  int tid = threadIdx.x;
  __shared__ float red[256];
  float ss = 0.f;
  for (int c = tid; c < C; c += 256) { float v = xr[c]; ss += v * v; }
  red[tid] = ss; __syncthreads();
  for (int o = 128; o > 0; o >>= 1) { if (tid < o) red[tid] += red[tid + o]; __syncthreads(); }
  float scale = rsqrtf(red[0] / (float)C + eps);
  for (int c = tid; c < C; c += 256) {
    float wv = w[c]; if (expw) wv = expf(wv);
    yr[c] = xr[c] * scale * wv;
  }
}

// out[r][c] = a[r][c] * silu(b[r][c])   (strided rows)
__global__ void mul_silu_kernel(float* __restrict__ o, const float* __restrict__ a,
                                const float* __restrict__ b,
                                int rows, int cols, int ldo, int lda, int ldb)
{
  long long i = (long long)blockIdx.x * blockDim.x + threadIdx.x;
  if (i >= (long long)rows * cols) return;
  int r = (int)(i / cols), c = (int)(i % cols);
  o[(size_t)r * ldo + c] = a[(size_t)r * lda + c] * siluf(b[(size_t)r * ldb + c]);
}

__global__ void silu_kernel(float* __restrict__ x, long long n)
{
  long long i = (long long)blockIdx.x * blockDim.x + threadIdx.x;
  if (i < n) x[i] = siluf(x[i]);
}

__global__ void add_kernel(float* __restrict__ o, const float* __restrict__ a,
                           const float* __restrict__ b, long long n)
{
  long long i = (long long)blockIdx.x * blockDim.x + threadIdx.x;
  if (i < n) o[i] = a[i] + b[i];
}

// causal depthwise conv (CK=4) + silu, xBC/out: (L, D)
__global__ void conv_silu_kernel(const float* __restrict__ xBC, const float* __restrict__ w,
                                 float* __restrict__ out, int L, int D)
{
  long long i = (long long)blockIdx.x * blockDim.x + threadIdx.x;
  if (i >= (long long)L * D) return;
  int l = (int)(i / D), c = (int)(i % D);
  float acc = 0.f;
  #pragma unroll
  for (int k = 0; k < 4; ++k) {
    int ls = l + k - 3;
    if (ls >= 0) acc += w[k * D + c] * xBC[(size_t)ls * D + c];
  }
  out[i] = siluf(acc);
}

// dt = softplus(dt_in + bias); dA = exp(dt * (-exp(A_log)))
__global__ void dt_kernel(const float* __restrict__ dt_in, const float* __restrict__ bias,
                          const float* __restrict__ A_log,
                          float* __restrict__ dt, float* __restrict__ dA, int L, int NG)
{
  int i = blockIdx.x * blockDim.x + threadIdx.x;
  if (i >= L * NG) return;
  int g = i % NG;
  float x = dt_in[i] + bias[g];
  float sp = (x > 20.f) ? x : log1pf(expf(x));
  float A = -expf(A_log[g]);
  dt[i] = sp;
  dA[i] = expf(sp * A);
}

// Selective scan: one block per group g. 256 threads; thread t holds state for
// h = t>>1 (0..127), s in [32*(t&1), 32*(t&1)+32). wave32 xor-shuffle combine.
__global__ __launch_bounds__(256) void ssm_scan_kernel(
    const float* __restrict__ conv,   // (L, 8192): xs | Bs | Cs
    const float* __restrict__ dt,     // (L,32)
    const float* __restrict__ dA,     // (L,32)
    const float* __restrict__ Df,     // (L,32)
    float* __restrict__ y,            // (L,4096)
    int L)
{
  const int g  = blockIdx.x;
  const int t  = threadIdx.x;
  const int h  = t >> 1;
  const int s0 = (t & 1) * 32;
  __shared__ float sx[128], sBv[64], sCv[64], ssc[3];
  float hs[32];
  #pragma unroll
  for (int j = 0; j < 32; ++j) hs[j] = 0.f;

  for (int l = 0; l < L; ++l) {
    const float* row = conv + (size_t)l * 8192;
    if (t < 128)       sx[t]        = row[g * 128 + t];
    else if (t < 192)  sBv[t - 128] = row[4096 + g * 64 + (t - 128)];
    else               sCv[t - 192] = row[6144 + g * 64 + (t - 192)];
    if (t == 0) { ssc[0] = dA[l * 32 + g]; ssc[1] = dt[l * 32 + g]; ssc[2] = Df[l * 32 + g]; }
    __syncthreads();
    float a = ssc[0], dtv = ssc[1], D = ssc[2];
    float xh = sx[h];
    float dx = dtv * xh;
    float acc = 0.f;
    #pragma unroll
    for (int j = 0; j < 32; ++j) {
      hs[j] = a * hs[j] + dx * sBv[s0 + j];
      acc += hs[j] * sCv[s0 + j];
    }
    acc += __shfl_xor(acc, 1, 32);
    if ((t & 1) == 0) y[(size_t)l * 4096 + g * 128 + h] = acc + D * xh;
    __syncthreads();
  }
}

// causal softmax over one score row; grid = (chunk_rows, heads)
__global__ __launch_bounds__(256) void softmax_kernel(float* __restrict__ sc,
                                                      int Lk, int qbase, int chunk)
{
  float* row = sc + ((size_t)blockIdx.y * chunk + blockIdx.x) * Lk;
  int valid = qbase + blockIdx.x + 1;
  int tid = threadIdx.x;
  __shared__ float red[256];
  float m = -3.4e38f;
  for (int c = tid; c < valid; c += 256) m = fmaxf(m, row[c]);
  red[tid] = m; __syncthreads();
  for (int o = 128; o > 0; o >>= 1) { if (tid < o) red[tid] = fmaxf(red[tid], red[tid + o]); __syncthreads(); }
  float mx = red[0];
  __syncthreads();
  float s = 0.f;
  for (int c = tid; c < valid; c += 256) { float e = expf(row[c] - mx); row[c] = e; s += e; }
  red[tid] = s; __syncthreads();
  for (int o = 128; o > 0; o >>= 1) { if (tid < o) red[tid] += red[tid + o]; __syncthreads(); }
  float inv = 1.f / red[0];
  for (int c = tid; c < Lk; c += 256) row[c] = (c < valid) ? row[c] * inv : 0.f;
}

// RoPE in place on (L, nh, 256); pair (i, i+128), theta=1e6
__global__ void rope_kernel(float* __restrict__ x, int L, int nh)
{
  long long idx = (long long)blockIdx.x * blockDim.x + threadIdx.x;
  if (idx >= (long long)L * nh * 128) return;
  int i = (int)(idx % 128);
  int t = (int)(idx / 128);
  int hh = t % nh, l = t / nh;
  float freq = __powf(1.0e6f, -(float)(2 * i) / 256.f);
  float ang = (float)l * freq;
  float s = sinf(ang), c = cosf(ang);
  size_t base = ((size_t)l * nh + hh) * 256;
  float x1 = x[base + i], x2 = x[base + i + 128];
  x[base + i]       = x1 * c - x2 * s;
  x[base + i + 128] = x2 * c + x1 * s;
}

// ---------------------------------------------------------------------------
// Host orchestration
// ---------------------------------------------------------------------------
extern "C" void kernel_launch(void* const* d_in, const int* in_sizes, int n_in,
                              void* d_out, int out_size, void* d_ws, size_t ws_size,
                              hipStream_t stream)
{
  (void)in_sizes; (void)n_in; (void)out_size; (void)ws_size;
  const int L = 2048, H = 2560, DI = 8192, DG = 4096, NG = 32, HDIM = 128;
  const int NH = 16, NKV = 4, HD = 256, INTER = 9216, OUTD = 1024;
  const int CH = 512;  // attention query chunk

  const float* x0       = (const float*)d_in[0];
  const float* s_inln   = (const float*)d_in[1];
  const float* s_zw     = (const float*)d_in[2];
  const float* s_qkvw   = (const float*)d_in[3];
  const float* s_bw     = (const float*)d_in[4];
  const float* s_aw     = (const float*)d_in[5];
  const float* s_convw  = (const float*)d_in[6];
  const float* s_outw   = (const float*)d_in[7];
  const float* s_normw  = (const float*)d_in[8];
  const float* s_Alog   = (const float*)d_in[9];
  const float* s_dtbias = (const float*)d_in[10];
  const float* s_postln = (const float*)d_in[11];
  const float* s_gatew  = (const float*)d_in[12];
  const float* s_upw    = (const float*)d_in[13];
  const float* s_downw  = (const float*)d_in[14];
  const float* a_inln   = (const float*)d_in[15];
  const float* a_qw     = (const float*)d_in[16];
  const float* a_kw     = (const float*)d_in[17];
  const float* a_vw     = (const float*)d_in[18];
  const float* a_ow     = (const float*)d_in[19];
  const float* a_qn     = (const float*)d_in[20];
  const float* a_kn     = (const float*)d_in[21];
  const float* a_postln = (const float*)d_in[22];
  const float* a_gatew  = (const float*)d_in[23];
  const float* a_upw    = (const float*)d_in[24];
  const float* a_downw  = (const float*)d_in[25];
  const float* h_w1     = (const float*)d_in[26];
  const float* h_expw   = (const float*)d_in[27];
  const float* h_w2     = (const float*)d_in[28];
  float* outp = (float*)d_out;

  char* wp = (char*)d_ws;
  auto alloc = [&](size_t n) -> float* {
    float* p = (float*)wp;
    wp += ((n * sizeof(float) + 255) & ~(size_t)255);
    return p;
  };
  float* xn   = alloc((size_t)L * H);      // normalized activations
  float* big0 = alloc((size_t)L * INTER);  // mlp gate
  float* big1 = alloc((size_t)L * INTER);  // mlp up / attention scores chunk
  float* big2 = alloc((size_t)L * DI);     // xBC / q+gate proj
  float* big3 = alloc((size_t)L * DI);     // conv out / gated attn out
  float* zb   = alloc((size_t)L * DG);     // z / attention O
  float* yb   = alloc((size_t)L * DG);     // scan y / gated y
  float* ynb  = alloc((size_t)L * DG);     // y norm / q compact (roped)
  float* co   = alloc((size_t)L * H);      // core / mlp projection out
  float* h1   = alloc((size_t)L * H);      // residual 1
  float* h2   = alloc((size_t)L * H);      // residual 2
  float* kraw = alloc((size_t)L * 1024);
  float* kc   = alloc((size_t)L * 1024);
  float* vb   = alloc((size_t)L * 1024);
  float* dinb = alloc((size_t)L * NG);
  float* Dbuf = alloc((size_t)L * NG);
  float* dtb  = alloc((size_t)L * NG);
  float* dAb  = alloc((size_t)L * NG);
  float* tb   = alloc((size_t)L * OUTD);
  float* tnb  = alloc((size_t)L * OUTD);

  auto gemm = [&](const float* A, const float* Bm, float* C, int M, int N, int K,
                  int lda, int ldb, int ldc, long long stA, long long stB, long long stC,
                  int divB, int batch, float alpha, bool transB) {
    dim3 grid((N + BN - 1) / BN, (M + BM - 1) / BM, batch);
    if (transB)
      gemm_wmma_kernel<true><<<grid, 256, 0, stream>>>(A, Bm, C, M, N, K, lda, ldb, ldc,
                                                       stA, stB, stC, divB, alpha);
    else
      gemm_wmma_kernel<false><<<grid, 256, 0, stream>>>(A, Bm, C, M, N, K, lda, ldb, ldc,
                                                        stA, stB, stC, divB, alpha);
  };
  auto rms = [&](const float* x, const float* w, float* y, int rows, int C, int ld,
                 int nh, int expw) {
    rmsnorm_kernel<<<rows, 256, 0, stream>>>(x, w, y, C, ld, nh, expw, 1e-6f);
  };
  auto mulsilu = [&](float* o, const float* a, const float* b, int rows, int cols,
                     int ldo, int lda, int ldb) {
    long long n = (long long)rows * cols;
    mul_silu_kernel<<<(unsigned)((n + 255) / 256), 256, 0, stream>>>(o, a, b, rows, cols, ldo, lda, ldb);
  };
  auto addv = [&](float* o, const float* a, const float* b, long long n) {
    add_kernel<<<(unsigned)((n + 255) / 256), 256, 0, stream>>>(o, a, b, n);
  };
  auto mlp = [&](const float* hin, const float* lnw, const float* gw, const float* uw,
                 const float* dw, float* hout) {
    rms(hin, lnw, xn, L, H, H, 1, 0);
    gemm(xn, gw, big0, L, INTER, H, H, INTER, INTER, 0, 0, 0, 1, 1, 1.f, false);
    gemm(xn, uw, big1, L, INTER, H, H, INTER, INTER, 0, 0, 0, 1, 1, 1.f, false);
    mulsilu(big0, big1, big0, L, INTER, INTER, INTER, INTER);   // up * silu(gate)
    gemm(big0, dw, co, L, H, INTER, INTER, H, H, 0, 0, 0, 1, 1, 1.f, false);
    addv(hout, hin, co, (long long)L * H);
  };

  // ================= SSM hybrid block =================
  rms(x0, s_inln, xn, L, H, H, 1, 0);
  gemm(xn, s_zw,   zb,   L, DG, H, H, DG, DG, 0, 0, 0, 1, 1, 1.f, false);
  gemm(xn, s_qkvw, big2, L, DI, H, H, DI, DI, 0, 0, 0, 1, 1, 1.f, false);
  gemm(xn, s_bw,   dinb, L, NG, H, H, NG, NG, 0, 0, 0, 1, 1, 1.f, false);
  gemm(xn, s_aw,   Dbuf, L, NG, H, H, NG, NG, 0, 0, 0, 1, 1, 1.f, false);
  conv_silu_kernel<<<(L * DI + 255) / 256, 256, 0, stream>>>(big2, s_convw, big3, L, DI);
  dt_kernel<<<(L * NG + 255) / 256, 256, 0, stream>>>(dinb, s_dtbias, s_Alog, dtb, dAb, L, NG);
  ssm_scan_kernel<<<NG, 256, 0, stream>>>(big3, dtb, dAb, Dbuf, yb, L);
  rms(yb, s_normw, ynb, L * NG, HDIM, HDIM, 1, 0);
  mulsilu(yb, ynb, zb, L, DG, DG, DG, DG);                      // y * silu(z)
  gemm(yb, s_outw, co, L, H, DG, DG, H, H, 0, 0, 0, 1, 1, 1.f, false);
  addv(h1, x0, co, (long long)L * H);
  mlp(h1, s_postln, s_gatew, s_upw, s_downw, h2);

  // ================= Attention hybrid block =================
  rms(h2, a_inln, xn, L, H, H, 1, 0);
  gemm(xn, a_qw, big2, L, 2 * NH * HD, H, H, 8192, 8192, 0, 0, 0, 1, 1, 1.f, false);
  gemm(xn, a_kw, kraw, L, NKV * HD, H, H, 1024, 1024, 0, 0, 0, 1, 1, 1.f, false);
  gemm(xn, a_vw, vb,   L, NKV * HD, H, H, 1024, 1024, 0, 0, 0, 1, 1, 1.f, false);
  rms(big2, a_qn, ynb, L * NH, HD, 8192, NH, 0);   // q-norm -> (L,4096) compact
  rms(kraw, a_kn, kc,  L * NKV, HD, 1024, NKV, 0); // k-norm -> (L,1024) compact
  rope_kernel<<<(L * NH * 128 + 255) / 256, 256, 0, stream>>>(ynb, L, NH);
  rope_kernel<<<(L * NKV * 128 + 255) / 256, 256, 0, stream>>>(kc, L, NKV);
  for (int c = 0; c < L / CH; ++c) {
    // scores = (Qc @ K^T) / 16 : batched over 16 heads, GQA divB=4
    gemm(ynb + (size_t)c * CH * 4096, kc, big1, CH, L, HD,
         4096, 1024, L, 256, 256, (long long)CH * L, NKV, NH, 0.0625f, true);
    softmax_kernel<<<dim3(CH, NH), 256, 0, stream>>>(big1, L, c * CH, CH);
    // O = att @ V
    gemm(big1, vb, zb + (size_t)c * CH * 4096, CH, HD, L,
         L, 1024, 4096, (long long)CH * L, 256, 256, NKV, NH, 1.f, false);
  }
  mulsilu(big3, zb, big2 + 4096, L, 4096, 4096, 4096, 8192);    // O * silu(gate)
  gemm(big3, a_ow, co, L, H, NH * HD, 4096, H, H, 0, 0, 0, 1, 1, 1.f, false);
  addv(h1, h2, co, (long long)L * H);
  mlp(h1, a_postln, a_gatew, a_upw, a_downw, h2);

  // ================= Head =================
  gemm(h2, h_w1, tb, L, OUTD, H, H, OUTD, OUTD, 0, 0, 0, 1, 1, 1.f, false);
  rms(tb, h_expw, tnb, L, OUTD, OUTD, 1, 1);                    // w = exp(exp_w)
  silu_kernel<<<(L * OUTD + 255) / 256, 256, 0, stream>>>(tnb, (long long)L * OUTD);
  gemm(tnb, h_w2, outp, L, OUTD, OUTD, OUTD, OUTD, OUTD, 0, 0, 0, 1, 1, 1.f, false);
}
